// HexConvResBlock_4964982194841
// MI455X (gfx1250) — matrix operand order repaired and checked by reading, not compiled, not run on hardware
//
#include <hip/hip_runtime.h>

// ---------------------------------------------------------------------------
// HexConvResBlock on MI455X (gfx1250, wave32, WMMA + async global->LDS)
//   conv as GEMM: M = B*165 = 168960, K = 7*256 = 1792, N = 256
//   bf16 inputs, f32 accumulate via v_wmma_f32_16x16x32_bf16
//   Block tile: M=64 x N=256, 8 waves, each wave: 2 M-subtiles x 4 N-subtiles
//   Double-buffered LDS; tiles staged with global_load_async_to_lds_b128;
//   one barrier + s_wait_asynccnt per K-step. OOB hex neighbors read a
//   zero-pad region (no EXEC-masked branches in the hot loop).
// ---------------------------------------------------------------------------

typedef __attribute__((ext_vector_type(16))) __bf16 v16bf;
typedef __attribute__((ext_vector_type(8)))  __bf16 v8bf;
typedef __attribute__((ext_vector_type(8)))  float  v8f;

#define GRID_HH 11
#define GRID_WW 15
#define NHEX    165
#define CCH     256
#define KTOT    1792          // 7 * 256
#define BATCH   1024
#define MTOT    (BATCH*NHEX)  // 168960
#define MTILE   64
#define KSTEP   32            // K per wmma (bf16)
#define NKSTEP  56            // 1792 / 32

// Packed (d+1) 2-bit tables for hex neighborhood j=0..6
#define DY_PACK  0x2950   // dy  = {-1,-1,0,0,0,1,1}
#define DXO_PACK 0x2649   // dxO = { 0, 1,-1,0,1,0,1}  (y odd)
#define DXE_PACK 0x1244   // dxE = {-1, 0,-1,0,1,-1,0} (y even)

__device__ __forceinline__ float leaky01(float v) { return v > 0.f ? v : 0.01f * v; }

// ---- CDNA5 async global->LDS staging (inline asm; ASYNCcnt-tracked) --------
__device__ __forceinline__ unsigned ldsAddrOf(const void* p)
{
    return (unsigned)(size_t)(__attribute__((address_space(3))) const char*)p;
}

__device__ __forceinline__ void asyncLoadB128(unsigned ldsDst, const void* gsrc)
{
    asm volatile("global_load_async_to_lds_b128 %0, %1, off"
                 :: "v"(ldsDst), "v"(gsrc) : "memory");
}

__device__ __forceinline__ void waitAsync0()
{
    asm volatile("s_wait_asynccnt 0x0" ::: "memory");
}

// --- convert W (1792x256, f32, K-major) -> Wt (256x1792, bf16, N-major) -----
__global__ __launch_bounds__(256)
void prep_weights(const float* __restrict__ W1, const float* __restrict__ W2,
                  __bf16* __restrict__ Wt1, __bf16* __restrict__ Wt2)
{
    int e = blockIdx.x * 256 + threadIdx.x;          // 0 .. 2*458752-1 (exact)
    const int per = KTOT * CCH;
    const float* W = W1;
    __bf16* Wt = Wt1;
    if (e >= per) { e -= per; W = W2; Wt = Wt2; }
    const int n = e / KTOT;
    const int k = e - n * KTOT;
    Wt[(size_t)n * KTOT + k] = (__bf16)W[(size_t)k * CCH + n];
}

// --- convert x (f32) -> xbf (bf16), and write the 512B zero-pad region ------
__global__ __launch_bounds__(256)
void prep_x(const float* __restrict__ x, __bf16* __restrict__ xbf,
            __bf16* __restrict__ zp)
{
    const size_t i = ((size_t)blockIdx.x * 256 + threadIdx.x) * 8;
    const float4 f0 = *(const float4*)(x + i);
    const float4 f1 = *(const float4*)(x + i + 4);
    union { uint4 u; __bf16 h[8]; } pk;
    pk.h[0] = (__bf16)f0.x; pk.h[1] = (__bf16)f0.y;
    pk.h[2] = (__bf16)f0.z; pk.h[3] = (__bf16)f0.w;
    pk.h[4] = (__bf16)f1.x; pk.h[5] = (__bf16)f1.y;
    pk.h[6] = (__bf16)f1.z; pk.h[7] = (__bf16)f1.w;
    *(uint4*)(xbf + i) = pk.u;

    if (blockIdx.x == 0 && threadIdx.x < 32) {       // 32 * 16B = 512B of zeros
        uint4 z; z.x = 0u; z.y = 0u; z.z = 0u; z.w = 0u;
        *(uint4*)((char*)zp + threadIdx.x * 16) = z;
    }
}

// --- fused gather + GEMM + bias + (residual) + leaky ------------------------
// SECOND=false: src = xbf (bf16), dst = h (bf16), no residual
// SECOND=true : src = h   (bf16), dst = out (f32), residual = x (f32)
template<bool SECOND>
__global__ __launch_bounds__(256)
void hexconv_kern(const __bf16* __restrict__ srcAct,
                  const __bf16* __restrict__ zeropad,  // >=512B of bf16 zeros
                  const __bf16* __restrict__ Wt,       // (256,1792) bf16
                  const float*  __restrict__ bias,     // (256,)
                  const float*  __restrict__ resid,    // x (f32) or nullptr
                  void*         __restrict__ dst)
{
    __shared__ __align__(16) __bf16 Atile[2][MTILE * KSTEP];  // 2 x 64x32 (8 KB)
    __shared__ __align__(16) __bf16 BT[2][CCH * KSTEP];       // 2 x 256x32 (32 KB)

    const int t     = threadIdx.x;
    const int w     = t >> 5;          // wave 0..7 (wave32)
    const int lane  = t & 31;
    const int half  = lane >> 4;
    const int ln    = lane & 15;
    const int mpair = w >> 2;          // 0..1 : M-subtiles {2*mpair, 2*mpair+1}
    const int nq    = w & 3;           // N-quadrant 0..3 (4 subtiles of 16)
    const int m0    = blockIdx.x * MTILE;

    // A-stage coords: 64 rows x 4 segments of 8 channels
    const int lr   = t >> 2;           // row in tile 0..63
    const int seg8 = t & 3;            // 0..3
    const int m    = m0 + lr;
    const int bb   = m / NHEX;
    const int p    = m - bb * NHEX;
    const int py   = p / GRID_WW;
    const int px   = p - py * GRID_WW;
    const bool yodd = (py & 1);

    // B-stage coords: thread t = output channel row of Wt (K-contiguous)
    const __bf16* wrow = Wt + (size_t)t * KTOT;

    // stage K-step ktn into buffer b: 5 async b128 per thread, branch-free
    auto stage = [&](int ktn, int b) {
        const int j  = ktn >> 3;
        const int kc = ktn & 7;
        const int dy = ((DY_PACK >> (2 * j)) & 3) - 1;
        const int dx = (((yodd ? DXO_PACK : DXE_PACK) >> (2 * j)) & 3) - 1;
        const int ny = py + dy, nx = px + dx;
        const bool ok = ((unsigned)ny < GRID_HH) && ((unsigned)nx < GRID_WW);
        const __bf16* abase =
            ok ? (srcAct + (long)(bb * NHEX + ny * GRID_WW + nx) * CCH) : zeropad;
        asyncLoadB128(ldsAddrOf(&Atile[b][lr * KSTEP + seg8 * 8]),
                      abase + kc * KSTEP + seg8 * 8);
        const __bf16* gB = wrow + (size_t)ktn * KSTEP;
        asyncLoadB128(ldsAddrOf(&BT[b][t * KSTEP]),     gB);
        asyncLoadB128(ldsAddrOf(&BT[b][t * KSTEP + 8]), gB + 8);
        asyncLoadB128(ldsAddrOf(&BT[b][t * KSTEP + 16]), gB + 16);
        asyncLoadB128(ldsAddrOf(&BT[b][t * KSTEP + 24]), gB + 24);
    };

    v8f acc[2][4];
#pragma unroll
    for (int mi = 0; mi < 2; ++mi)
#pragma unroll
        for (int q = 0; q < 4; ++q)
            acc[mi][q] = (v8f){0.f,0.f,0.f,0.f,0.f,0.f,0.f,0.f};

    // ---- prologue: stage K-step 0 into buffer 0 ----
    stage(0, 0);
    waitAsync0();
    __syncthreads();

#pragma unroll 2
    for (int kt = 0; kt < NKSTEP; ++kt) {
        const int buf = kt & 1;

        // fire async staging for K-step kt+1 into the other buffer
        if ((kt + 1) < NKSTEP) stage(kt + 1, buf ^ 1);

        // ---- fragments + 8 x wmma from buffer `buf` ----
        const __bf16* Ab = &Atile[buf][0];
        const __bf16* Bb = &BT[buf][0];

        v16bf af[2];
#pragma unroll
        for (int mi = 0; mi < 2; ++mi) {
            const __bf16* ar = Ab + (mpair * 32 + mi * 16 + ln) * KSTEP + half * 8;
            const v8bf a0 = *(const v8bf*)(ar);
            const v8bf a1 = *(const v8bf*)(ar + 16);
            af[mi] = __builtin_shufflevector(a0, a1,
                        0,1,2,3,4,5,6,7,8,9,10,11,12,13,14,15);
        }
#pragma unroll
        for (int q = 0; q < 4; ++q) {
            const __bf16* br = Bb + ((nq * 4 + q) * 16 + ln) * KSTEP + half * 8;
            const v8bf b0 = *(const v8bf*)(br);
            const v8bf b1 = *(const v8bf*)(br + 16);
            const v16bf bfb = __builtin_shufflevector(b0, b1,
                                0,1,2,3,4,5,6,7,8,9,10,11,12,13,14,15);
#pragma unroll
            for (int mi = 0; mi < 2; ++mi)
                acc[mi][q] = __builtin_amdgcn_wmma_f32_16x16x32_bf16(
                                 false, af[mi], false, bfb, (short)0,
                                 acc[mi][q], false, false);
        }

        // own async stores must land before the group-wide barrier
        waitAsync0();
        __syncthreads();
    }

    // ---- epilogue: bias (+residual) + leaky, store ----
    const int nbase = nq * 64;
#pragma unroll
    for (int mi = 0; mi < 2; ++mi) {
#pragma unroll
        for (int q = 0; q < 4; ++q) {
            const int n  = nbase + q * 16 + ln;
            const float bv = bias[n];
#pragma unroll
            for (int r = 0; r < 8; ++r) {
                const int row = m0 + mpair * 32 + mi * 16 + half * 8 + r;
                const size_t idx = (size_t)row * CCH + n;
                float v = acc[mi][q][r] + bv;
                if (SECOND) v += resid[idx];
                v = leaky01(v);
                if (SECOND) ((float*)dst)[idx]  = v;
                else        ((__bf16*)dst)[idx] = (__bf16)v;
            }
        }
    }
}

// ---------------------------------------------------------------------------
extern "C" void kernel_launch(void* const* d_in, const int* in_sizes, int n_in,
                              void* d_out, int out_size, void* d_ws, size_t ws_size,
                              hipStream_t stream)
{
    const float* x  = (const float*)d_in[0];
    const float* W1 = (const float*)d_in[1];
    const float* b1 = (const float*)d_in[2];
    const float* W2 = (const float*)d_in[3];
    const float* b2 = (const float*)d_in[4];
    float* out = (float*)d_out;

    char* ws = (char*)d_ws;
    const size_t wtBytes  = (size_t)CCH * KTOT * 2;          // 917,504 B each
    const size_t actBytes = (size_t)MTOT * CCH * 2;          // 86,507,520 B each
    __bf16* Wt1 = (__bf16*)(ws);
    __bf16* Wt2 = (__bf16*)(ws + wtBytes);
    __bf16* xbf = (__bf16*)(ws + 2 * wtBytes);
    __bf16* h   = (__bf16*)(ws + 2 * wtBytes + actBytes);
    __bf16* zp  = (__bf16*)(ws + 2 * wtBytes + 2 * actBytes); // 512 B zeros

    prep_weights<<<3584, 256, 0, stream>>>(W1, W2, Wt1, Wt2);
    prep_x<<<(MTOT * CCH) / (256 * 8), 256, 0, stream>>>(x, xbf, zp);  // 21120 blocks

    const int nblk = MTOT / MTILE;                           // 2640
    hexconv_kern<false><<<nblk, 256, 0, stream>>>(xbf, zp, Wt1, b1, nullptr, h);
    hexconv_kern<true ><<<nblk, 256, 0, stream>>>(h,   zp, Wt2, b2, x, out);
}
